// layer_68186900791806
// MI455X (gfx1250) — compile-verified, matrix operand order checked by reading, not run on hardware
//
#include <hip/hip_runtime.h>

// ---------------------------------------------------------------------------
// Fused MLP: out1 = mean(leaky(x@W1+b1)), out2 = mean(leaky(h1@W2+b2))
// MI455X (gfx1250): wave32, WMMA 16x16x32 f16->f32, 320KB LDS/WGP.
//
// Roofline: 23.6 GFLOP vs 212 MB HBM traffic -> f16 WMMA puts us at the
// 23.3 TB/s memory roof (~9us); f32 WMMA (K=4) would be compute-bound.
// Accumulation is f32; f16 only on A/B operands -> ~1e-3 rel error on h,
// averaged down by the 150/100-wide row means. Fully fused: h1 lives in LDS.
// ---------------------------------------------------------------------------

typedef __attribute__((ext_vector_type(16))) _Float16 v16h;
typedef __attribute__((ext_vector_type(8)))  _Float16 v8h;
typedef __attribute__((ext_vector_type(8)))  float    v8f;

#define LEAKY_ALPHA 0.1f

// Problem dims (from reference)
constexpr int D_IN = 200;
constexpr int D1   = 150;
constexpr int D2   = 100;

// Tiling / padding
constexpr int ROWS     = 128;  // rows per workgroup (8 waves x 16)
constexpr int K1_STEPS = 7;    // 7*32 = 224 >= 200
constexpr int KP1      = 232;  // LDS row stride (halves): 116 dwords, gcd-free banks
constexpr int NT1      = 10;   // 10*16 = 160 >= 150
constexpr int NP1      = 160;
constexpr int K2_STEPS = 5;    // 5*32 = 160 >= 150
constexpr int KP2      = 168;  // LDS row stride (halves): 84 dwords
constexpr int NT2      = 7;    // 7*16 = 112 >= 100
constexpr int NP2      = 112;

constexpr int LDS_X  = ROWS * KP1;   // x tile  (f16)
constexpr int LDS_W1 = NP1  * KP1;   // W1^T    (f16)
constexpr int LDS_H1 = ROWS * KP2;   // h1 tile (f16)
constexpr int LDS_W2 = NP2  * KP2;   // W2^T    (f16)
constexpr int LDS_HALVES = LDS_X + LDS_W1 + LDS_H1 + LDS_W2;   // 107136 -> 214272 B

__global__ __launch_bounds__(256, 1)
void mlp2_fused_wmma(const float* __restrict__ x,
                     const float* __restrict__ W1,
                     const float* __restrict__ b1,
                     const float* __restrict__ W2,
                     const float* __restrict__ b2,
                     float* __restrict__ out,   // [0,B): out1  [B,2B): out2
                     int Btotal)
{
    extern __shared__ _Float16 smem[];
    _Float16* xlds  = smem;
    _Float16* w1lds = xlds  + LDS_X;
    _Float16* h1lds = w1lds + LDS_W1;
    _Float16* w2lds = h1lds + LDS_H1;

    const int tid     = threadIdx.x;
    const int wave    = tid >> 5;
    const int lane    = tid & 31;
    const int m       = lane & 15;   // lane's M (for A) and N (for B/C column)
    const int hh      = lane >> 4;   // half-wave selector in WMMA layouts
    const int rowBase = blockIdx.x * ROWS;
    const int rowL    = wave * 16 + m;   // this lane's local A row

    // ---------------- Stage weights (transposed over K) and x tile ---------
    // W1^T: w1lds[n*KP1 + k] = W1[k*D1 + n]  (zero padded)
    for (int idx = tid; idx < NP1 * KP1; idx += 256) {
        const int n = idx % NP1;          // consecutive threads -> consecutive n
        const int k = idx / NP1;          //   => coalesced global reads
        float v = (n < D1 && k < D_IN) ? W1[k * D1 + n] : 0.f;
        w1lds[n * KP1 + k] = (_Float16)v;
    }
    // W2^T: w2lds[n*KP2 + k] = W2[k*D2 + n]
    for (int idx = tid; idx < NP2 * KP2; idx += 256) {
        const int n = idx % NP2;
        const int k = idx / NP2;
        float v = (n < D2 && k < D1) ? W2[k * D2 + n] : 0.f;
        w2lds[n * KP2 + k] = (_Float16)v;
    }
    // x tile: xlds[r*KP1 + k] = x[(rowBase+r)*D_IN + k]   (coalesced along k)
    for (int idx = tid; idx < ROWS * KP1; idx += 256) {
        const int k = idx % KP1;
        const int r = idx / KP1;
        const int gr = rowBase + r;
        float v = (k < D_IN && gr < Btotal) ? x[(size_t)gr * D_IN + k] : 0.f;
        xlds[r * KP1 + k] = (_Float16)v;
    }
    __syncthreads();

    // ---------------- Layer 1: h1 = leaky(x@W1 + b1), row partial sums -----
    float rs1[8];
#pragma unroll
    for (int v = 0; v < 8; ++v) rs1[v] = 0.f;

    for (int nt = 0; nt < NT1; ++nt) {
        v8f acc = {};
        const int ncol = nt * 16 + m;                         // C column of this lane
        const _Float16* aBase = xlds  + rowL * KP1 + 8 * hh;
        const _Float16* bBase = w1lds + ncol * KP1 + 8 * hh;
#pragma unroll
        for (int ks = 0; ks < K1_STEPS; ++ks) {
            union { v16h v; v8h h[2]; } a, b;
            a.h[0] = *(const v8h*)(aBase + ks * 32);       // K = 32ks+8hh+[0..7]
            a.h[1] = *(const v8h*)(aBase + ks * 32 + 16);  // K = 32ks+16+8hh+[0..7]
            b.h[0] = *(const v8h*)(bBase + ks * 32);
            b.h[1] = *(const v8h*)(bBase + ks * 32 + 16);
            acc = __builtin_amdgcn_wmma_f32_16x16x32_f16(
                false, a.v, false, b.v, (short)0, acc, false, false);
        }
        const float bias = (ncol < D1) ? b1[ncol] : 0.f;
#pragma unroll
        for (int v = 0; v < 8; ++v) {
            float z  = acc[v] + bias;
            float hv = (z >= 0.f) ? z : LEAKY_ALPHA * z;
            if (ncol >= D1) hv = 0.f;                      // padded column -> exact 0
            rs1[v] += hv;
            // C layout: VGPR v, lane -> row v + 8*hh, col ncol
            h1lds[(wave * 16 + v + 8 * hh) * KP2 + ncol] = (_Float16)hv;
        }
    }

    // Row mean out1: reduce the 16 lanes of each half (cols 0..159, zeros padded)
#pragma unroll
    for (int v = 0; v < 8; ++v) {
        float s = rs1[v];
        s += __shfl_xor(s, 1, 32);
        s += __shfl_xor(s, 2, 32);
        s += __shfl_xor(s, 4, 32);
        s += __shfl_xor(s, 8, 32);
        if (m == 0) {
            const int gr = rowBase + wave * 16 + v + 8 * hh;
            if (gr < Btotal) out[gr] = s * (1.0f / (float)D1);
        }
    }

    __syncthreads();  // belt-and-braces; layer-2 only reads this wave's own h1 rows

    // ---------------- Layer 2: out2 = mean(leaky(h1@W2 + b2)) --------------
    float rs2[8];
#pragma unroll
    for (int v = 0; v < 8; ++v) rs2[v] = 0.f;

    for (int nt = 0; nt < NT2; ++nt) {
        v8f acc = {};
        const int ncol = nt * 16 + m;
        const _Float16* aBase = h1lds + rowL * KP2 + 8 * hh;
        const _Float16* bBase = w2lds + ncol * KP2 + 8 * hh;
#pragma unroll
        for (int ks = 0; ks < K2_STEPS; ++ks) {
            union { v16h v; v8h h[2]; } a, b;
            a.h[0] = *(const v8h*)(aBase + ks * 32);
            a.h[1] = *(const v8h*)(aBase + ks * 32 + 16);
            b.h[0] = *(const v8h*)(bBase + ks * 32);
            b.h[1] = *(const v8h*)(bBase + ks * 32 + 16);
            acc = __builtin_amdgcn_wmma_f32_16x16x32_f16(
                false, a.v, false, b.v, (short)0, acc, false, false);
        }
        const float bias = (ncol < D2) ? b2[ncol] : 0.f;
#pragma unroll
        for (int v = 0; v < 8; ++v) {
            float z  = acc[v] + bias;
            float hv = (z >= 0.f) ? z : LEAKY_ALPHA * z;
            if (ncol >= D2) hv = 0.f;
            rs2[v] += hv;
        }
    }

#pragma unroll
    for (int v = 0; v < 8; ++v) {
        float s = rs2[v];
        s += __shfl_xor(s, 1, 32);
        s += __shfl_xor(s, 2, 32);
        s += __shfl_xor(s, 4, 32);
        s += __shfl_xor(s, 8, 32);
        if (m == 0) {
            const int gr = rowBase + wave * 16 + v + 8 * hh;
            if (gr < Btotal) out[(size_t)Btotal + gr] = s * (1.0f / (float)D2);
        }
    }
}

extern "C" void kernel_launch(void* const* d_in, const int* in_sizes, int n_in,
                              void* d_out, int out_size, void* d_ws, size_t ws_size,
                              hipStream_t stream) {
    const float* x  = (const float*)d_in[0];
    const float* W1 = (const float*)d_in[1];
    const float* b1 = (const float*)d_in[2];
    const float* W2 = (const float*)d_in[3];
    const float* b2 = (const float*)d_in[4];
    float* out = (float*)d_out;

    const int Btotal = in_sizes[0] / D_IN;         // 262144
    const int grid   = (Btotal + ROWS - 1) / ROWS; // 2048
    const size_t shmem = (size_t)LDS_HALVES * sizeof(_Float16); // ~214 KB (< 320 KB/WGP)

    hipLaunchKernelGGL(mlp2_fused_wmma, dim3(grid), dim3(256), shmem, stream,
                       x, W1, b1, W2, b2, out, Btotal);
}
